// InformationFusion_34248069218980
// MI455X (gfx1250) — matrix operand order repaired and checked by reading, not compiled, not run on hardware
//
#include <hip/hip_runtime.h>
#include <hip/hip_bf16.h>
#include <math.h>

// ---------------------------------------------------------------------------
// MI455X (gfx1250, wave32): LDS-tiled, double-buffered, software-pipelined
// bf16 WMMA GEMM pipeline.
// Block = 256 threads (8 waves). Block tile 128(M) x 64(N), K step 32.
// Each wave computes a 32x32 sub-tile = 2x2 v_wmma_f32_16x16x32_bf16 accs.
// LDS rows padded to 40 bf16 (80B, 16B-aligned) -> conflict-free b128 reads.
// Pipeline: global loads for stage i+1 are issued right after the barrier and
// only waited on in the NEXT iteration's LDS-store phase, so HBM/L2 latency
// overlaps the 4 WMMAs + ds_loads of the current stage.
// ---------------------------------------------------------------------------

typedef __attribute__((ext_vector_type(16))) __bf16 v16bf;
typedef __attribute__((ext_vector_type(4)))  __bf16 v4bf;
typedef __attribute__((ext_vector_type(8)))  float  v8f;

#define LDSTR 40   // padded LDS row stride in bf16 elements (32 data + 8 pad)

__device__ __forceinline__ v8f wmma_bf16(v16bf a, v16bf b, v8f c) {
    return __builtin_amdgcn_wmma_f32_16x16x32_bf16(false, a, false, b,
                                                   (short)0, c, false, false);
}

// Branch-free tanh: 1 - 2/(exp(2x)+1). exp overflow->inf => 1; exp->0 => -1.
__device__ __forceinline__ float fast_tanh(float x) {
    const float e = __expf(2.0f * x);
    return 1.0f - __fdividef(2.0f, e + 1.0f);
}

__device__ __forceinline__ v4bf cvt4(float4 v) {
    v4bf o; o[0] = (__bf16)v.x; o[1] = (__bf16)v.y; o[2] = (__bf16)v.z; o[3] = (__bf16)v.w;
    return o;
}

__device__ __forceinline__ v16bf pack16(v4bf x0, v4bf x1, v4bf x2, v4bf x3) {
    v16bf r;
#pragma unroll
    for (int i = 0; i < 4; ++i) { r[i] = x0[i]; r[4+i] = x1[i]; r[8+i] = x2[i]; r[12+i] = x3[i]; }
    return r;
}

// ---- staged registers (global -> regs -> LDS, split so loads batch) ----

struct RA_F { float4 v[4]; };   // A tile stage, f32 source (128x32 / 256 thr)
struct RA_H { v4bf  v[4]; };    // A tile stage, bf16 form
struct RB_F { float4 v[2]; };   // B tile stage, f32 NT source (64x32)
struct RB_H { v4bf  v[2]; };    // B tile stage, bf16 NT source
struct RB_N { float v[8]; };    // B tile stage, f32 NN (transposing) source

// A: thread i-th vector -> row r = vec>>3 (0..127), kc = (vec&7)*4
__device__ __forceinline__ RA_F loadA_f32(const float* __restrict__ A, int lda,
                                          int row0, int k0, int tid) {
    RA_F rg;
#pragma unroll
    for (int i = 0; i < 4; ++i) {
        const int vec = tid + i * 256, r = vec >> 3, kc = (vec & 7) << 2;
        rg.v[i] = *(const float4*)(A + (size_t)(row0 + r) * lda + k0 + kc);
    }
    return rg;
}
__device__ __forceinline__ void storeA_f32(__bf16* lA, const RA_F& rg, int tid) {
#pragma unroll
    for (int i = 0; i < 4; ++i) {
        const int vec = tid + i * 256, r = vec >> 3, kc = (vec & 7) << 2;
        *(v4bf*)(lA + r * LDSTR + kc) = cvt4(rg.v[i]);
    }
}

__device__ __forceinline__ RA_H loadA_bf16(const __bf16* __restrict__ A, int lda,
                                           int row0, int k0, int tid) {
    RA_H rg;
#pragma unroll
    for (int i = 0; i < 4; ++i) {
        const int vec = tid + i * 256, r = vec >> 3, kc = (vec & 7) << 2;
        rg.v[i] = *(const v4bf*)(A + (size_t)(row0 + r) * lda + k0 + kc);
    }
    return rg;
}
__device__ __forceinline__ void storeA_h(__bf16* lA, const RA_H& rg, int tid) {
#pragma unroll
    for (int i = 0; i < 4; ++i) {
        const int vec = tid + i * 256, r = vec >> 3, kc = (vec & 7) << 2;
        *(v4bf*)(lA + r * LDSTR + kc) = rg.v[i];
    }
}

// mixed A source for the concat GEMM: k<768 from f32 target, else bf16 weighted
__device__ __forceinline__ RA_H loadA_mix(const float* __restrict__ F,
                                          const __bf16* __restrict__ H,
                                          int row0, int k0, int tid) {
    RA_H rg;
    if (k0 < 768) {
#pragma unroll
        for (int i = 0; i < 4; ++i) {
            const int vec = tid + i * 256, r = vec >> 3, kc = (vec & 7) << 2;
            rg.v[i] = cvt4(*(const float4*)(F + (size_t)(row0 + r) * 768 + k0 + kc));
        }
    } else {
#pragma unroll
        for (int i = 0; i < 4; ++i) {
            const int vec = tid + i * 256, r = vec >> 3, kc = (vec & 7) << 2;
            rg.v[i] = *(const v4bf*)(H + (size_t)(row0 + r) * 768 + (k0 - 768) + kc);
        }
    }
    return rg;
}

// B (NT): lB[n][k] = W[n0+n][k0+k]; n = vec>>3 (0..63), kc = (vec&7)*4
__device__ __forceinline__ RB_F loadB_nt_f32(const float* __restrict__ W, int ldk,
                                             int n0, int k0, int tid) {
    RB_F rg;
#pragma unroll
    for (int i = 0; i < 2; ++i) {
        const int vec = tid + i * 256, n = vec >> 3, kc = (vec & 7) << 2;
        rg.v[i] = *(const float4*)(W + (size_t)(n0 + n) * ldk + k0 + kc);
    }
    return rg;
}
__device__ __forceinline__ void storeB_f32(__bf16* lB, const RB_F& rg, int tid) {
#pragma unroll
    for (int i = 0; i < 2; ++i) {
        const int vec = tid + i * 256, n = vec >> 3, kc = (vec & 7) << 2;
        *(v4bf*)(lB + n * LDSTR + kc) = cvt4(rg.v[i]);
    }
}

__device__ __forceinline__ RB_H loadB_nt_bf16(const __bf16* __restrict__ W, int ldk,
                                              int n0, int k0, int tid) {
    RB_H rg;
#pragma unroll
    for (int i = 0; i < 2; ++i) {
        const int vec = tid + i * 256, n = vec >> 3, kc = (vec & 7) << 2;
        rg.v[i] = *(const v4bf*)(W + (size_t)(n0 + n) * ldk + k0 + kc);
    }
    return rg;
}
__device__ __forceinline__ void storeB_h(__bf16* lB, const RB_H& rg, int tid) {
#pragma unroll
    for (int i = 0; i < 2; ++i) {
        const int vec = tid + i * 256, n = vec >> 3, kc = (vec & 7) << 2;
        *(v4bf*)(lB + n * LDSTR + kc) = rg.v[i];
    }
}

// B (NN transposing): lB[n][k] = S[k0+k][n0+n]; e = tid+i*256, k=e>>6, n=e&63
__device__ __forceinline__ RB_N loadB_nn_f32(const float* __restrict__ S, int ldn,
                                             int k0, int n0, int tid) {
    RB_N rg;
#pragma unroll
    for (int i = 0; i < 8; ++i) {
        const int e = tid + i * 256, k = e >> 6, n = e & 63;
        rg.v[i] = S[(size_t)(k0 + k) * ldn + n0 + n];
    }
    return rg;
}
__device__ __forceinline__ void storeB_nn(__bf16* lB, const RB_N& rg, int tid) {
#pragma unroll
    for (int i = 0; i < 8; ++i) {
        const int e = tid + i * 256, k = e >> 6, n = e & 63;
        lB[n * LDSTR + k] = (__bf16)rg.v[i];
    }
}

// ---- fragment loads from LDS (ISA 7.12.2 bf16 layouts) ----

// A fragment 16x32: lane m = lane&15, elems 0..7 <-> K=8h+i, 8..15 <-> K=16+8h+i
__device__ __forceinline__ v16bf fragA(const __bf16* lA, int mt, int lane) {
    const int m = (mt << 4) + (lane & 15);
    const int h = lane >> 4;
    const __bf16* p = lA + m * LDSTR + (h << 3);
    return pack16(*(const v4bf*)(p),      *(const v4bf*)(p + 4),
                  *(const v4bf*)(p + 16), *(const v4bf*)(p + 20));
}

// B fragment 32x16: lane n = lane&15, elems 0..7 <-> K=8h+i, 8..15 <-> K=16+8h+i
__device__ __forceinline__ v16bf fragB(const __bf16* lB, int nt, int lane) {
    const int n = (nt << 4) + (lane & 15);
    const int h = lane >> 4;
    const __bf16* p = lB + n * LDSTR + (h << 3);
    return pack16(*(const v4bf*)(p),      *(const v4bf*)(p + 4),
                  *(const v4bf*)(p + 16), *(const v4bf*)(p + 20));
}

// 2x2 wave-tile compute step from one LDS stage
__device__ __forceinline__ void mma_step(const __bf16* lA, const __bf16* lB,
                                         int wm, int wn, int lane, v8f acc[2][2]) {
    v16bf a0 = fragA(lA, 2 * wm, lane);
    v16bf a1 = fragA(lA, 2 * wm + 1, lane);
    v16bf b0 = fragB(lB, 2 * wn, lane);
    v16bf b1 = fragB(lB, 2 * wn + 1, lane);
    acc[0][0] = wmma_bf16(a0, b0, acc[0][0]);
    acc[0][1] = wmma_bf16(a0, b1, acc[0][1]);
    acc[1][0] = wmma_bf16(a1, b0, acc[1][0]);
    acc[1][1] = wmma_bf16(a1, b1, acc[1][1]);
}

// ===========================================================================
// Kernel 1: H = tanh(A @ W^T + b), A f32 [M,K], W f32 [N,K], out bf16 [M,N]
// ===========================================================================
__global__ __launch_bounds__(256) void k_lin_tanh(const float* __restrict__ A,
                                                  const float* __restrict__ W,
                                                  const float* __restrict__ bias,
                                                  __bf16* __restrict__ out,
                                                  int M, int N, int K) {
    __shared__ __bf16 lA[2][128 * LDSTR];
    __shared__ __bf16 lB[2][64 * LDSTR];
    const int tid = threadIdx.x;
    const int lane = tid & 31, wave = tid >> 5;
    const int wm = wave >> 1, wn = wave & 1;
    const int bn = N >> 6;
    const int row0 = (blockIdx.x / bn) << 7;
    const int n0   = (blockIdx.x % bn) << 6;

    v8f acc[2][2] = {};
    RA_F ra = loadA_f32(A, K, row0, 0, tid);
    RB_F rb = loadB_nt_f32(W, K, n0, 0, tid);
    int cur = 0;
    for (int k0 = 0; k0 < K; k0 += 32) {
        storeA_f32(lA[cur], ra, tid);
        storeB_f32(lB[cur], rb, tid);
        __syncthreads();
        if (k0 + 32 < K) {
            ra = loadA_f32(A, K, row0, k0 + 32, tid);
            rb = loadB_nt_f32(W, K, n0, k0 + 32, tid);
        }
        mma_step(lA[cur], lB[cur], wm, wn, lane, acc);
        cur ^= 1;
    }
    const int nl = lane & 15, h = lane >> 4;
#pragma unroll
    for (int im = 0; im < 2; ++im)
#pragma unroll
        for (int in = 0; in < 2; ++in) {
            const int n = n0 + ((2 * wn + in) << 4) + nl;
            const float bv = bias[n];
#pragma unroll
            for (int v = 0; v < 8; ++v) {
                const int m = row0 + ((2 * wm + im) << 4) + v + 8 * h;
                out[(size_t)m * N + n] = (__bf16)fast_tanh(acc[im][in][v] + bv);
            }
        }
}

// ===========================================================================
// Kernel 2: scores[b] = target_h[b] @ aux_h[b]^T   (both bf16 [1024,512])
// ===========================================================================
__global__ __launch_bounds__(256) void k_scores(const __bf16* __restrict__ th,
                                                const __bf16* __restrict__ ah,
                                                __bf16* __restrict__ sc) {
    __shared__ __bf16 lA[2][128 * LDSTR];
    __shared__ __bf16 lB[2][64 * LDSTR];
    const int b = blockIdx.y;
    const __bf16* A = th + (size_t)b * 1024 * 512;
    const __bf16* W = ah + (size_t)b * 1024 * 512;
    __bf16* out = sc + (size_t)b * 1024 * 1024;

    const int tid = threadIdx.x;
    const int lane = tid & 31, wave = tid >> 5;
    const int wm = wave >> 1, wn = wave & 1;
    const int row0 = (blockIdx.x >> 4) << 7;   // 8 row-blocks x 16 col-blocks
    const int n0   = (blockIdx.x & 15) << 6;

    v8f acc[2][2] = {};
    RA_H ra = loadA_bf16(A, 512, row0, 0, tid);
    RB_H rb = loadB_nt_bf16(W, 512, n0, 0, tid);
    int cur = 0;
    for (int k0 = 0; k0 < 512; k0 += 32) {
        storeA_h(lA[cur], ra, tid);
        storeB_h(lB[cur], rb, tid);
        __syncthreads();
        if (k0 + 32 < 512) {
            ra = loadA_bf16(A, 512, row0, k0 + 32, tid);
            rb = loadB_nt_bf16(W, 512, n0, k0 + 32, tid);
        }
        mma_step(lA[cur], lB[cur], wm, wn, lane, acc);
        cur ^= 1;
    }
    const int nl = lane & 15, h = lane >> 4;
#pragma unroll
    for (int im = 0; im < 2; ++im)
#pragma unroll
        for (int in = 0; in < 2; ++in) {
            const int n = n0 + ((2 * wn + in) << 4) + nl;
#pragma unroll
            for (int v = 0; v < 8; ++v) {
                const int m = row0 + ((2 * wm + im) << 4) + v + 8 * h;
                out[(size_t)m * 1024 + n] = (__bf16)acc[im][in][v];
            }
        }
}

// ===========================================================================
// Kernel 3: column softmax over t (axis=1) of scores[b][t][a], in place.
// ===========================================================================
__global__ __launch_bounds__(256) void k_softmax_col(__bf16* __restrict__ sc) {
    const int b = blockIdx.y;
    const int a = blockIdx.x * blockDim.x + threadIdx.x;   // 0..1023
    __bf16* col = sc + (size_t)b * 1024 * 1024 + a;
    float mx = -3.0e38f;
    for (int t = 0; t < 1024; ++t) mx = fmaxf(mx, (float)col[(size_t)t * 1024]);
    float s = 0.0f;
    for (int t = 0; t < 1024; ++t) s += __expf((float)col[(size_t)t * 1024] - mx);
    const float inv = 1.0f / s;
    for (int t = 0; t < 1024; ++t) {
        const float v = __expf((float)col[(size_t)t * 1024] - mx) * inv;
        col[(size_t)t * 1024] = (__bf16)v;
    }
}

// ===========================================================================
// Kernel 4: weighted[b] = attn[b] @ aux[b]   (attn bf16 [1024,1024], aux f32 NN)
// ===========================================================================
__global__ __launch_bounds__(256) void k_weighted(const __bf16* __restrict__ sc,
                                                  const float* __restrict__ aux,
                                                  __bf16* __restrict__ wg) {
    __shared__ __bf16 lA[2][128 * LDSTR];
    __shared__ __bf16 lB[2][64 * LDSTR];
    const int b = blockIdx.y;
    const __bf16* A = sc + (size_t)b * 1024 * 1024;
    const float* S = aux + (size_t)b * 1024 * 768;
    __bf16* out = wg + (size_t)b * 1024 * 768;

    const int tid = threadIdx.x;
    const int lane = tid & 31, wave = tid >> 5;
    const int wm = wave >> 1, wn = wave & 1;
    const int row0 = (blockIdx.x / 12) << 7;   // 8 row-blocks x 12 col-blocks
    const int n0   = (blockIdx.x % 12) << 6;

    v8f acc[2][2] = {};
    RA_H ra = loadA_bf16(A, 1024, row0, 0, tid);
    RB_N rb = loadB_nn_f32(S, 768, 0, n0, tid);
    int cur = 0;
    for (int k0 = 0; k0 < 1024; k0 += 32) {
        storeA_h(lA[cur], ra, tid);
        storeB_nn(lB[cur], rb, tid);
        __syncthreads();
        if (k0 + 32 < 1024) {
            ra = loadA_bf16(A, 1024, row0, k0 + 32, tid);
            rb = loadB_nn_f32(S, 768, k0 + 32, n0, tid);
        }
        mma_step(lA[cur], lB[cur], wm, wn, lane, acc);
        cur ^= 1;
    }
    const int nl = lane & 15, h = lane >> 4;
#pragma unroll
    for (int im = 0; im < 2; ++im)
#pragma unroll
        for (int in = 0; in < 2; ++in) {
            const int n = n0 + ((2 * wn + in) << 4) + nl;
#pragma unroll
            for (int v = 0; v < 8; ++v) {
                const int m = row0 + ((2 * wm + im) << 4) + v + 8 * h;
                out[(size_t)m * 768 + n] = (__bf16)acc[im][in][v];
            }
        }
}

// ===========================================================================
// Kernel 5: out[b][o][t] = tanh(concat(target, weighted)[b][t] . Wf[o] + bf[o])
// M = 8192 (batch folded), N = 768, K = 1536 (k<768: f32 target; else bf16 wg)
// ===========================================================================
__global__ __launch_bounds__(256) void k_final(const float* __restrict__ target,
                                               const __bf16* __restrict__ wg,
                                               const float* __restrict__ Wf,
                                               const float* __restrict__ bf_,
                                               float* __restrict__ out) {
    __shared__ __bf16 lA[2][128 * LDSTR];
    __shared__ __bf16 lB[2][64 * LDSTR];
    const int tid = threadIdx.x;
    const int lane = tid & 31, wave = tid >> 5;
    const int wm = wave >> 1, wn = wave & 1;
    const int row0 = (blockIdx.x / 12) << 7;   // 64 row-blocks x 12 col-blocks
    const int n0   = (blockIdx.x % 12) << 6;

    v8f acc[2][2] = {};
    RA_H ra = loadA_mix(target, wg, row0, 0, tid);
    RB_F rb = loadB_nt_f32(Wf, 1536, n0, 0, tid);
    int cur = 0;
    for (int k0 = 0; k0 < 1536; k0 += 32) {
        storeA_h(lA[cur], ra, tid);
        storeB_f32(lB[cur], rb, tid);
        __syncthreads();
        if (k0 + 32 < 1536) {
            ra = loadA_mix(target, wg, row0, k0 + 32, tid);
            rb = loadB_nt_f32(Wf, 1536, n0, k0 + 32, tid);
        }
        mma_step(lA[cur], lB[cur], wm, wn, lane, acc);
        cur ^= 1;
    }
    const int nl = lane & 15, h = lane >> 4;
#pragma unroll
    for (int im = 0; im < 2; ++im)
#pragma unroll
        for (int in = 0; in < 2; ++in) {
            const int n = n0 + ((2 * wn + in) << 4) + nl;
            const float bv = bf_[n];
#pragma unroll
            for (int v = 0; v < 8; ++v) {
                const int m = row0 + ((2 * wm + im) << 4) + v + 8 * h; // = b*1024+t
                const int bb_ = m >> 10;
                const int t = m & 1023;
                out[(size_t)bb_ * 768 * 1024 + (size_t)n * 1024 + t] =
                    fast_tanh(acc[im][in][v] + bv);
            }
        }
}

// ===========================================================================
extern "C" void kernel_launch(void* const* d_in, const int* in_sizes, int n_in,
                              void* d_out, int out_size, void* d_ws, size_t ws_size,
                              hipStream_t stream) {
    const float* target = (const float*)d_in[0];   // [8,1024,768]
    const float* aux    = (const float*)d_in[1];   // [8,1024,768]
    const float* Wt     = (const float*)d_in[2];   // [512,768]
    const float* bt     = (const float*)d_in[3];   // [512]
    const float* Wa     = (const float*)d_in[4];   // [512,768]
    const float* ba     = (const float*)d_in[5];   // [512]
    const float* Wf     = (const float*)d_in[6];   // [768,1536]
    const float* bf_    = (const float*)d_in[7];   // [768]
    float* out = (float*)d_out;                    // [8,768,1024]

    // Workspace layout (32 MB total):
    //   [0 , 8MB)  target_h bf16 [8,1024,512]   -- later reused for weighted
    //   [8 ,16MB)  aux_h    bf16 [8,1024,512]
    //   [16,32MB)  scores/attn bf16 [8,1024,1024] (softmaxed in place)
    char* ws = (char*)d_ws;
    __bf16* th = (__bf16*)ws;
    __bf16* ah = th + (size_t)8 * 1024 * 512;
    __bf16* sc = (__bf16*)(ws + (size_t)16 * 1024 * 1024);
    __bf16* wg = (__bf16*)ws;   // reuse region 0 after scores are done

    dim3 blk(256);
    // 1,2: hidden projections (M=8192,N=512,K=768): (8192/128)*(512/64)=512 blocks
    k_lin_tanh<<<512, blk, 0, stream>>>(target, Wt, bt, th, 8192, 512, 768);
    k_lin_tanh<<<512, blk, 0, stream>>>(aux,    Wa, ba, ah, 8192, 512, 768);
    // 3: scores per batch (M=N=1024,K=512): 8*16=128 blocks per batch
    k_scores<<<dim3(128, 8), blk, 0, stream>>>(th, ah, sc);
    // 4: softmax over t for each (b, a) column
    k_softmax_col<<<dim3(4, 8), blk, 0, stream>>>(sc);
    // 5: weighted aux per batch (M=1024,N=768,K=1024): 8*12=96 blocks per batch
    k_weighted<<<dim3(96, 8), blk, 0, stream>>>(sc, aux, wg);
    // 6: fused concat + final projection + tanh + transpose: 64*12=768 blocks
    k_final<<<768, blk, 0, stream>>>(target, wg, Wf, bf_, out);
}
